// AttentionGoalState_51762945851647
// MI455X (gfx1250) — matrix operand order repaired and checked by reading, not compiled, not run on hardware
//
#include <hip/hip_runtime.h>
#include <hip/hip_bf16.h>
#include <math.h>

// ---------------------------------------------------------------------------
// CDNA5 (gfx1250) wave32 WMMA bf16 GEMM implementation of the reference net.
// All heavy math runs through v_wmma_f32_16x16x32_bf16 (bf16 A/B, f32 acc).
// Inner GEMM loops are branch-free (no divergent guards around WMMA; EXEC
// stays all-ones as required by the ISA) and software-pipelined 2-deep.
// ---------------------------------------------------------------------------

typedef __attribute__((ext_vector_type(16))) __bf16 v16b;
typedef __attribute__((ext_vector_type(8)))  __bf16 v8b;
typedef __attribute__((ext_vector_type(8)))  float  v8f;

#define WMMA_BF16(a, b, c) \
  __builtin_amdgcn_wmma_f32_16x16x32_bf16(false, (a), false, (b), (short)0, (c), false, false)

// A fragment (16x32 bf16): lane holds row m=lane&15; element j <-> k =
// (lane>>4)*16 + j  => two 16B loads at +0 and +8 (ptr pre-
// offset by (lane>>4)*16).
__device__ __forceinline__ v16b ldA(const __bf16* p) {
  union { v16b v; v8b h[2]; } u;
  u.h[0] = *reinterpret_cast<const v8b*>(p);
  u.h[1] = *reinterpret_cast<const v8b*>(p + 8);
  return u.v;
}
// B fragment (32x16 bf16, operand stored transposed [N][K]): lane holds col
// n=lane&15; element j <-> k = (lane>>4)*16 + j => contiguous 32B chunk (ptr
// pre-offset by (lane>>4)*16).
__device__ __forceinline__ v16b ldB(const __bf16* p) {
  union { v16b v; v8b h[2]; } u;
  u.h[0] = *reinterpret_cast<const v8b*>(p);
  u.h[1] = *reinterpret_cast<const v8b*>(p + 8);
  return u.v;
}

// ---------------------------------------------------------------------------
// Generic batched GEMM: C[M][N] = A[M][K] * Bt[N][K]^T  (+bias[m], opt relu)
// grid: (ceil(N/64), ceil(M/64), batch); block: 128 (4 waves, 16 M-rows each)
// ---------------------------------------------------------------------------
template <bool HAS_BIAS, bool RELU>
__global__ __launch_bounds__(128)
void gemm_nt(const __bf16* __restrict__ A, long aB, int lda,
             const __bf16* __restrict__ Bt, long bB, int ldb,
             float* __restrict__ C, long cB, int ldc,
             const float* __restrict__ bias,
             int M, int N, int K)
{
  const int z = blockIdx.z;
  A  += (long)z * aB;
  Bt += (long)z * bB;
  C  += (long)z * cB;
  const int wave = threadIdx.x >> 5, lane = threadIdx.x & 31;
  const int m0 = (blockIdx.y * 4 + wave) * 16;
  if (m0 >= M) return;                       // uniform per wave
  const int n0 = blockIdx.x * 64;
  const int lrow = lane & 15;
  const int hA = (lane >> 4) << 4;
  const int hB = (lane >> 4) << 4;
  // Invalid N-tiles are clamped to tile 0 (always exists): WMMA runs
  // unconditionally, dead results are dropped in the epilogue.
  const bool nv1 = (n0 + 16) < N, nv2 = (n0 + 32) < N, nv3 = (n0 + 48) < N;
  const __bf16* ap  = A + (long)(m0 + lrow) * lda + hA;
  const __bf16* bp0 = Bt + (long)(n0 + lrow) * ldb + hB;
  const __bf16* bp1 = Bt + (long)((nv1 ? n0 + 16 : 0) + lrow) * ldb + hB;
  const __bf16* bp2 = Bt + (long)((nv2 ? n0 + 32 : 0) + lrow) * ldb + hB;
  const __bf16* bp3 = Bt + (long)((nv3 ? n0 + 48 : 0) + lrow) * ldb + hB;

  v8f acc0 = {}, acc1 = {}, acc2 = {}, acc3 = {};
  v16b aF = ldA(ap);
  v16b b0 = ldB(bp0), b1 = ldB(bp1), b2 = ldB(bp2), b3 = ldB(bp3);
  for (int k0 = 0; k0 < K; k0 += 32) {
    const int adv = (k0 + 32 < K) ? 32 : 0;  // scalar select; last iter reloads
    ap += adv; bp0 += adv; bp1 += adv; bp2 += adv; bp3 += adv;
    v16b aN = ldA(ap);
    v16b n0f = ldB(bp0), n1f = ldB(bp1), n2f = ldB(bp2), n3f = ldB(bp3);
    acc0 = WMMA_BF16(aF, b0, acc0);
    acc1 = WMMA_BF16(aF, b1, acc1);
    acc2 = WMMA_BF16(aF, b2, acc2);
    acc3 = WMMA_BF16(aF, b3, acc3);
    aF = aN; b0 = n0f; b1 = n1f; b2 = n2f; b3 = n3f;
  }

  const int mBase = m0 + ((lane >> 4) << 3);
  v8f accs[4] = {acc0, acc1, acc2, acc3};
  const bool nvs[4] = {true, nv1, nv2, nv3};
#pragma unroll
  for (int c = 0; c < 4; ++c) {
    if (!nvs[c]) continue;
    const int n = n0 + c * 16 + lrow;
#pragma unroll
    for (int r = 0; r < 8; ++r) {
      float v = accs[c][r];
      const int m = mBase + r;
      if (HAS_BIAS) v += bias[m];
      if (RELU) v = fmaxf(v, 0.f);
      C[(long)m * ldc + n] = v;
    }
  }
}

// ---------------------------------------------------------------------------
// Temporal Conv3d k=3 VALID as 27-tap implicit GEMM over xT[b][s][1024] bf16.
// M=128 (out ch), N=400 (4*10*10 positions), K=1024 per tap.
// out = relu(conv + bias) -> gconv[b][128][400]
// ---------------------------------------------------------------------------
__global__ __launch_bounds__(128)
void tconv_gemm(const __bf16* __restrict__ Wt,   // [27][128][1024]
                const __bf16* __restrict__ xT,   // [8][864][1024]
                const float* __restrict__ bias,
                float* __restrict__ out)         // [8][128][400]
{
  const int b = blockIdx.z;
  const int wave = threadIdx.x >> 5, lane = threadIdx.x & 31;
  const int m0 = (blockIdx.y * 4 + wave) * 16;
  if (m0 >= 128) return;
  const int n0 = blockIdx.x * 64;
  const int lrow = lane & 15;
  const int hA = (lane >> 4) << 4, hB = (lane >> 4) << 4;
  const __bf16* xb = xT + (long)b * 864 * 1024;
  bool nv[4];
  int pc[4];
#pragma unroll
  for (int c = 0; c < 4; ++c) {
    nv[c] = (n0 + c * 16) < 400;
    pc[c] = (nv[c] ? (n0 + c * 16) : 0) + lrow;   // clamp dead tiles to tile 0
  }
  v8f acc0 = {}, acc1 = {}, acc2 = {}, acc3 = {};
  for (int tap = 0; tap < 27; ++tap) {
    const int dt = tap / 9, dh = (tap % 9) / 3, dw = tap % 3;
    const __bf16* ap = Wt + ((long)tap * 128 + m0 + lrow) * 1024 + hA;
    const __bf16* bp[4];
#pragma unroll
    for (int c = 0; c < 4; ++c) {
      const int p = pc[c];
      const int tp = p / 100, q = p % 100, hp = q / 10, wp = q % 10;
      const int s = (tp + dt) * 144 + (hp + dh) * 12 + (wp + dw);
      bp[c] = xb + (long)s * 1024 + hB;
    }
    const __bf16 *bp0 = bp[0], *bp1 = bp[1], *bp2 = bp[2], *bp3 = bp[3];
    v16b aF = ldA(ap);
    v16b b0 = ldB(bp0), b1 = ldB(bp1), b2 = ldB(bp2), b3 = ldB(bp3);
    for (int k0 = 0; k0 < 1024; k0 += 32) {
      const int adv = (k0 + 32 < 1024) ? 32 : 0;
      ap += adv; bp0 += adv; bp1 += adv; bp2 += adv; bp3 += adv;
      v16b aN = ldA(ap);
      v16b n0f = ldB(bp0), n1f = ldB(bp1), n2f = ldB(bp2), n3f = ldB(bp3);
      acc0 = WMMA_BF16(aF, b0, acc0);
      acc1 = WMMA_BF16(aF, b1, acc1);
      acc2 = WMMA_BF16(aF, b2, acc2);
      acc3 = WMMA_BF16(aF, b3, acc3);
      aF = aN; b0 = n0f; b1 = n1f; b2 = n2f; b3 = n3f;
    }
  }
  const int mBase = m0 + ((lane >> 4) << 3);
  v8f accs[4] = {acc0, acc1, acc2, acc3};
#pragma unroll
  for (int c = 0; c < 4; ++c) {
    if (!nv[c]) continue;
    const int p = n0 + c * 16 + lrow;
#pragma unroll
    for (int r = 0; r < 8; ++r) {
      const int m = mBase + r;
      out[((long)b * 128 + m) * 400 + p] = fmaxf(accs[c][r] + bias[m], 0.f);
    }
  }
}

// ---------------------------------------------------------------------------
// ConvTranspose2d #1 (1024->512, k=2, s=2) : per (d,k) GEMM, epilogue writes
// relu(.+b) as bf16 directly into the transposed layout h1T[32*24*24][512].
// N = 4608 and M = 512 are exact multiples of the tile -> no guards at all.
// ---------------------------------------------------------------------------
__global__ __launch_bounds__(128)
void up1_gemm(const __bf16* __restrict__ A,     // [4][512][1024]
              const __bf16* __restrict__ Bt,    // [4608][1024] = frame^T
              const float* __restrict__ bias,
              __bf16* __restrict__ h1T)         // [18432][512]
{
  const int dk = blockIdx.z, d = dk >> 1, kk = dk & 1;
  const int wave = threadIdx.x >> 5, lane = threadIdx.x & 31;
  const int m0 = (blockIdx.y * 4 + wave) * 16;
  const int n0 = blockIdx.x * 64;
  const int lrow = lane & 15;
  const int hA = (lane >> 4) << 4, hB = (lane >> 4) << 4;
  const __bf16* ap  = A + (long)dk * 512 * 1024 + (long)(m0 + lrow) * 1024 + hA;
  const __bf16* bp0 = Bt + (long)(n0 + lrow) * 1024 + hB;
  const __bf16* bp1 = bp0 + 16 * 1024;
  const __bf16* bp2 = bp0 + 32 * 1024;
  const __bf16* bp3 = bp0 + 48 * 1024;
  v8f acc0 = {}, acc1 = {}, acc2 = {}, acc3 = {};
  v16b aF = ldA(ap);
  v16b b0 = ldB(bp0), b1 = ldB(bp1), b2 = ldB(bp2), b3 = ldB(bp3);
  for (int k0 = 0; k0 < 1024; k0 += 32) {
    const int adv = (k0 + 32 < 1024) ? 32 : 0;
    ap += adv; bp0 += adv; bp1 += adv; bp2 += adv; bp3 += adv;
    v16b aN = ldA(ap);
    v16b n0f = ldB(bp0), n1f = ldB(bp1), n2f = ldB(bp2), n3f = ldB(bp3);
    acc0 = WMMA_BF16(aF, b0, acc0);
    acc1 = WMMA_BF16(aF, b1, acc1);
    acc2 = WMMA_BF16(aF, b2, acc2);
    acc3 = WMMA_BF16(aF, b3, acc3);
    aF = aN; b0 = n0f; b1 = n1f; b2 = n2f; b3 = n3f;
  }
  const int mBase = m0 + ((lane >> 4) << 3);
  v8f accs[4] = {acc0, acc1, acc2, acc3};
#pragma unroll
  for (int c = 0; c < 4; ++c) {
    const int n1 = n0 + c * 16 + lrow;
    const int nn = n1 / 144, hw = n1 % 144, hh = hw / 12, w = hw % 12;
    const long row = (long)nn * 576 + (2 * hh + d) * 24 + (2 * w + kk);
#pragma unroll
    for (int r = 0; r < 8; ++r) {
      const int m = mBase + r;
      h1T[row * 512 + m] = (__bf16)fmaxf(accs[c][r] + bias[m], 0.f);
    }
  }
}

// ---------------------------------------------------------------------------
// ConvTranspose2d #2 (512->128, k=2, s=2) : epilogue writes f32 state_embed^T
// h2T[n][s=48x48][128]  (s-major so the final attention is coalesced).
// ---------------------------------------------------------------------------
__global__ __launch_bounds__(128)
void up2_gemm(const __bf16* __restrict__ A,     // [4][128][512]
              const __bf16* __restrict__ Bt,    // [18432][512] = h1T
              const float* __restrict__ bias,
              float* __restrict__ h2T)          // [32][2304][128]
{
  const int dk = blockIdx.z, d = dk >> 1, kk = dk & 1;
  const int wave = threadIdx.x >> 5, lane = threadIdx.x & 31;
  const int m0 = (blockIdx.y * 4 + wave) * 16;
  const int n0 = blockIdx.x * 64;
  const int lrow = lane & 15;
  const int hA = (lane >> 4) << 4, hB = (lane >> 4) << 4;
  const __bf16* ap  = A + (long)dk * 128 * 512 + (long)(m0 + lrow) * 512 + hA;
  const __bf16* bp0 = Bt + (long)(n0 + lrow) * 512 + hB;
  const __bf16* bp1 = bp0 + 16 * 512;
  const __bf16* bp2 = bp0 + 32 * 512;
  const __bf16* bp3 = bp0 + 48 * 512;
  v8f acc0 = {}, acc1 = {}, acc2 = {}, acc3 = {};
  v16b aF = ldA(ap);
  v16b b0 = ldB(bp0), b1 = ldB(bp1), b2 = ldB(bp2), b3 = ldB(bp3);
  for (int k0 = 0; k0 < 512; k0 += 32) {
    const int adv = (k0 + 32 < 512) ? 32 : 0;
    ap += adv; bp0 += adv; bp1 += adv; bp2 += adv; bp3 += adv;
    v16b aN = ldA(ap);
    v16b n0f = ldB(bp0), n1f = ldB(bp1), n2f = ldB(bp2), n3f = ldB(bp3);
    acc0 = WMMA_BF16(aF, b0, acc0);
    acc1 = WMMA_BF16(aF, b1, acc1);
    acc2 = WMMA_BF16(aF, b2, acc2);
    acc3 = WMMA_BF16(aF, b3, acc3);
    aF = aN; b0 = n0f; b1 = n1f; b2 = n2f; b3 = n3f;
  }
  const int mBase = m0 + ((lane >> 4) << 3);
  v8f accs[4] = {acc0, acc1, acc2, acc3};
#pragma unroll
  for (int c = 0; c < 4; ++c) {
    const int n2 = n0 + c * 16 + lrow;
    const int nn = n2 / 576, rem = n2 % 576, H = rem / 24, W = rem % 24;
    const long s = (2 * H + d) * 48 + (2 * W + kk);
#pragma unroll
    for (int r = 0; r < 8; ++r) {
      const int m = mBase + r;
      h2T[((long)nn * 2304 + s) * 128 + m] = accs[c][r] + bias[m];
    }
  }
}

// ------------------------- data-movement helpers ---------------------------

__global__ void cvt_bf16(const float* __restrict__ in, __bf16* __restrict__ out, long total) {
  long i = (long)blockIdx.x * 256 + threadIdx.x;
  if (i < total) out[i] = (__bf16)in[i];
}

// context [8][6][1024][144] -> x f32 [8][1024][864] and xT bf16 [8][864][1024]
__global__ void prep_context(const float* __restrict__ ctx, float* __restrict__ xF,
                             __bf16* __restrict__ xT, long total) {
  long i = (long)blockIdx.x * 256 + threadIdx.x;
  if (i >= total) return;
  const int s = (int)(i % 864); const long t1 = i / 864;
  const int c = (int)(t1 % 1024), b = (int)(t1 / 1024);
  const int tt = s / 144, hw = s % 144;
  const float v = ctx[(((long)b * 6 + tt) * 1024 + c) * 144 + hw];
  xF[i] = v;
  xT[((long)b * 864 + s) * 1024 + c] = (__bf16)v;
}

// frame [8][4][1024][144] -> frT bf16 [4608][1024]
__global__ void prep_frame(const float* __restrict__ fr, __bf16* __restrict__ frT, long total) {
  long i = (long)blockIdx.x * 256 + threadIdx.x;
  if (i >= total) return;
  const int hw = (int)(i % 144); const long t1 = i / 144;
  const int c = (int)(t1 % 1024), n = (int)(t1 / 1024);
  frT[((long)n * 144 + hw) * 1024 + c] = (__bf16)fr[i];
}

// tp_w [128][1024][27] -> [27][128][1024] bf16
__global__ void pack_tpw(const float* __restrict__ in, __bf16* __restrict__ out, long total) {
  long i = (long)blockIdx.x * 256 + threadIdx.x;
  if (i >= total) return;
  const int tap = (int)(i % 27); const long t = i / 27;
  const int c = (int)(t % 1024), o = (int)(t / 1024);
  out[((long)tap * 128 + o) * 1024 + c] = (__bf16)in[i];
}

// up1_w [1024][512][2][2] -> [4][512][1024] bf16
__global__ void pack_up1(const float* __restrict__ in, __bf16* __restrict__ out, long total) {
  long i = (long)blockIdx.x * 256 + threadIdx.x;
  if (i >= total) return;
  const int dk = (int)(i % 4); const long t = i / 4;
  const int o = (int)(t % 512), c = (int)(t / 512);
  out[((long)dk * 512 + o) * 1024 + c] = (__bf16)in[i];
}

// up2_w [512][128][2][2] -> [4][128][512] bf16
__global__ void pack_up2(const float* __restrict__ in, __bf16* __restrict__ out, long total) {
  long i = (long)blockIdx.x * 256 + threadIdx.x;
  if (i >= total) return;
  const int dk = (int)(i % 4); const long t = i / 4;
  const int o = (int)(t % 128), c = (int)(t / 128);
  out[((long)dk * 128 + o) * 512 + c] = (__bf16)in[i];
}

// [B][C][S] f32 -> [B][S][C] bf16
__global__ void t_cs(const float* __restrict__ in, __bf16* __restrict__ out,
                     int C, int S, long total) {
  long i = (long)blockIdx.x * 256 + threadIdx.x;
  if (i >= total) return;
  const int s = (int)(i % S); const long t = i / S;
  const int c = (int)(t % C), b = (int)(t / C);
  out[((long)b * S + s) * C + c] = (__bf16)in[i];
}

// Vn^T = bf16(relu(V[b][c][s] * rowsum[b][s]))  -> [B][S][512]
__global__ void vn_transpose(const float* __restrict__ V, const float* __restrict__ rs,
                             __bf16* __restrict__ out, long total) {
  long i = (long)blockIdx.x * 256 + threadIdx.x;
  if (i >= total) return;
  const int s = (int)(i % 864); const long t = i / 864;
  const int c = (int)(t % 512), b = (int)(t / 512);
  const float v = V[i] * rs[(long)b * 864 + s];
  out[((long)b * 864 + s) * 512 + c] = (__bf16)fmaxf(v, 0.f);
}

__global__ void residual_add(float* __restrict__ x, const float* __restrict__ o, long total) {
  long i = (long)blockIdx.x * 256 + threadIdx.x;
  if (i < total) x[i] += o[i];
}

// faithful softmax-rowsum: rs[row] = sum_t softmax(row*scale)[t]  (~1.0)
__global__ __launch_bounds__(256)
void rowsum_softmax(const float* __restrict__ KQ, float* __restrict__ rs, int S, float scale) {
  const long row = blockIdx.x;
  const float* p = KQ + row * (long)S;
  __shared__ float red[256];
  const int tid = threadIdx.x;
  float mx = -3.4e38f;
  for (int t = tid; t < S; t += 256) mx = fmaxf(mx, p[t] * scale);
  red[tid] = mx; __syncthreads();
  for (int st = 128; st > 0; st >>= 1) { if (tid < st) red[tid] = fmaxf(red[tid], red[tid + st]); __syncthreads(); }
  mx = red[0]; __syncthreads();
  float se = 0.f;
  for (int t = tid; t < S; t += 256) se += __expf(p[t] * scale - mx);
  red[tid] = se; __syncthreads();
  for (int st = 128; st > 0; st >>= 1) { if (tid < st) red[tid] += red[tid + st]; __syncthreads(); }
  const float inv = 1.f / red[0]; __syncthreads();
  float acc = 0.f;
  for (int t = tid; t < S; t += 256) acc += __expf(p[t] * scale - mx) * inv;
  red[tid] = acc; __syncthreads();
  for (int st = 128; st > 0; st >>= 1) { if (tid < st) red[tid] += red[tid + st]; __syncthreads(); }
  if (tid == 0) rs[row] = red[0];
}

// BatchNorm3d (training stats) over (B,S) per channel; normalizes y in place
// and emits the transposed bf16 copy for the next GEMM stage.
__global__ __launch_bounds__(256)
void bn_fused(float* __restrict__ y, __bf16* __restrict__ xT,
              const float* __restrict__ gamma, const float* __restrict__ beta,
              int B, int C, int S) {
  const int c = blockIdx.x;
  __shared__ float s1[256], s2[256];
  const int tid = threadIdx.x;
  const int tot = B * S;
  float a = 0.f, b2 = 0.f;
  for (int i = tid; i < tot; i += 256) {
    const int b = i / S, s = i % S;
    const float v = y[((long)b * C + c) * S + s];
    a += v; b2 += v * v;
  }
  s1[tid] = a; s2[tid] = b2; __syncthreads();
  for (int st = 128; st > 0; st >>= 1) {
    if (tid < st) { s1[tid] += s1[tid + st]; s2[tid] += s2[tid + st]; }
    __syncthreads();
  }
  const float mean = s1[0] / tot;
  const float var = s2[0] / tot - mean * mean;
  const float inv = rsqrtf(var + 1e-5f);
  const float g = gamma[c], be = beta[c];
  for (int i = tid; i < tot; i += 256) {
    const int b = i / S, s = i % S;
    const long idx = ((long)b * C + c) * S + s;
    const float v = (y[idx] - mean) * inv * g + be;
    y[idx] = v;
    xT[((long)b * S + s) * C + c] = (__bf16)v;
  }
}

__global__ __launch_bounds__(128)
void pool_goal(const float* __restrict__ gconv, float* __restrict__ out) {
  const int b = blockIdx.x, o = threadIdx.x;
  const float* p = gconv + ((long)b * 128 + o) * 400;
  float s = 0.f;
  for (int i = 0; i < 400; ++i) s += p[i];
  out[b * 128 + o] = s * (1.f / 400.f);
}

// per-n goal/state attention weights (softmax over s); torch .repeat => n%B
__global__ __launch_bounds__(256)
void sg_softmax(const float* __restrict__ h2T, const float* __restrict__ goal,
                float* __restrict__ attnW) {
  const int n = blockIdx.x, tid = threadIdx.x;
  const float* h2 = h2T + (long)n * 2304 * 128;
  float* aw = attnW + (long)n * 2304;
  __shared__ float g[128];
  __shared__ float red[256];
  if (tid < 128) g[tid] = goal[(n & 7) * 128 + tid];
  __syncthreads();
  const float scale = 0.0883883476483184f;  // 1/sqrt(128)
  float mx = -3.4e38f;
  for (int s = tid; s < 2304; s += 256) {
    float dot = 0.f;
    for (int d = 0; d < 128; ++d) dot += h2[(long)s * 128 + d] * g[d];
    const float sg = dot * scale;
    aw[s] = sg;
    mx = fmaxf(mx, sg);
  }
  red[tid] = mx; __syncthreads();
  for (int st = 128; st > 0; st >>= 1) { if (tid < st) red[tid] = fmaxf(red[tid], red[tid + st]); __syncthreads(); }
  mx = red[0]; __syncthreads();
  float se = 0.f;
  for (int s = tid; s < 2304; s += 256) { const float e = __expf(aw[s] - mx); aw[s] = e; se += e; }
  red[tid] = se; __syncthreads();
  for (int st = 128; st > 0; st >>= 1) { if (tid < st) red[tid] += red[tid + st]; __syncthreads(); }
  const float inv = 1.f / red[0];
  for (int s = tid; s < 2304; s += 256) aw[s] *= inv;
}

// out[n][d] = sum_s attn[n][s] * (state_embed^T[n][s][d] + pe[s][d])
__global__ __launch_bounds__(128)
void attn_out(const float* __restrict__ h2T, const float* __restrict__ attnW,
              float* __restrict__ out) {
  const int n = blockIdx.x, d = threadIdx.x;
  const float* h2 = h2T + (long)n * 2304 * 128;
  const float* aw = attnW + (long)n * 2304;
  const float f = __expf(-(float)(d & ~1) * (9.21034037197618f / 128.f));  // 10000^(-2i/D)
  float acc = 0.f;
  for (int s = 0; s < 2304; ++s) {
    const float pe = (d & 1) ? __cosf(s * f) : __sinf(s * f);
    acc += aw[s] * (h2[(long)s * 128 + d] + pe);
  }
  out[1024 + n * 128 + d] = acc;
}

// ---------------------------------------------------------------------------

extern "C" void kernel_launch(void* const* d_in, const int* in_sizes, int n_in,
                              void* d_out, int out_size, void* d_ws, size_t ws_size,
                              hipStream_t stream) {
  (void)in_sizes; (void)n_in; (void)out_size; (void)ws_size;
  const float* ctx   = (const float*)d_in[0];
  const float* frame = (const float*)d_in[1];
  const float* Kw = (const float*)d_in[2];
  const float* Kb = (const float*)d_in[3];
  const float* Qw = (const float*)d_in[4];
  const float* Qb = (const float*)d_in[5];
  const float* Vw = (const float*)d_in[6];
  const float* Vb = (const float*)d_in[7];
  const float* Ow = (const float*)d_in[8];
  const float* Ob = (const float*)d_in[9];
  const float* gamma = (const float*)d_in[10];
  const float* beta  = (const float*)d_in[11];
  const float* tpw = (const float*)d_in[12];
  const float* tpb = (const float*)d_in[13];
  const float* u1w = (const float*)d_in[14];
  const float* u1b = (const float*)d_in[15];
  const float* u2w = (const float*)d_in[16];
  const float* u2b = (const float*)d_in[17];
  float* out = (float*)d_out;

  const int B = 8, C = 1024, Cf = 512, S = 864;
  char* ws = (char*)d_ws;
  size_t cur = 0;
  auto alloc = [&](size_t bytes) -> void* {
    void* p = ws + cur;
    cur = (cur + bytes + 255) & ~(size_t)255;
    return p;
  };

  float*  xF    = (float*)alloc((size_t)B * C * S * 4);        // x (residual stream)
  __bf16* xT    = (__bf16*)alloc((size_t)B * S * C * 2);       // x^T bf16
  float*  Kf    = (float*)alloc((size_t)B * Cf * S * 4);
  float*  Qf    = (float*)alloc((size_t)B * Cf * S * 4);
  float*  Vf    = (float*)alloc((size_t)B * Cf * S * 4);
  __bf16* KtB   = (__bf16*)alloc((size_t)B * S * Cf * 2);
  __bf16* QtB   = (__bf16*)alloc((size_t)B * S * Cf * 2);
  float*  KQ    = (float*)alloc((size_t)B * S * S * 4);
  float*  RS    = (float*)alloc((size_t)B * S * 4);
  __bf16* VntB  = (__bf16*)alloc((size_t)B * S * Cf * 2);
  float*  Of    = (float*)alloc((size_t)B * C * S * 4);
  __bf16* wKb   = (__bf16*)alloc((size_t)2 * Cf * C * 2);
  __bf16* wQb   = (__bf16*)alloc((size_t)2 * Cf * C * 2);
  __bf16* wVb   = (__bf16*)alloc((size_t)2 * Cf * C * 2);
  __bf16* wOb   = (__bf16*)alloc((size_t)2 * C * Cf * 2);
  __bf16* tpwB  = (__bf16*)alloc((size_t)27 * 128 * 1024 * 2);
  __bf16* up1A  = (__bf16*)alloc((size_t)4 * 512 * 1024 * 2);
  __bf16* up2A  = (__bf16*)alloc((size_t)4 * 128 * 512 * 2);
  __bf16* frT   = (__bf16*)alloc((size_t)4608 * 1024 * 2);
  __bf16* h1T   = (__bf16*)alloc((size_t)18432 * 512 * 2);
  float*  h2T   = (float*)alloc((size_t)32 * 2304 * 128 * 4);
  float*  gconv = (float*)alloc((size_t)8 * 128 * 400 * 4);
  float*  attnW = (float*)alloc((size_t)32 * 2304 * 4);

  auto blocks = [](long n) { return (unsigned)((n + 255) / 256); };

  // ---- operand conversions / packing ----
  const long wlen = 2L * Cf * C;
  cvt_bf16<<<blocks(wlen), 256, 0, stream>>>(Kw, wKb, wlen);
  cvt_bf16<<<blocks(wlen), 256, 0, stream>>>(Qw, wQb, wlen);
  cvt_bf16<<<blocks(wlen), 256, 0, stream>>>(Vw, wVb, wlen);
  cvt_bf16<<<blocks(wlen), 256, 0, stream>>>(Ow, wOb, wlen);
  pack_tpw<<<blocks(128L * 1024 * 27), 256, 0, stream>>>(tpw, tpwB, 128L * 1024 * 27);
  pack_up1<<<blocks(1024L * 512 * 4), 256, 0, stream>>>(u1w, up1A, 1024L * 512 * 4);
  pack_up2<<<blocks(512L * 128 * 4), 256, 0, stream>>>(u2w, up2A, 512L * 128 * 4);
  prep_frame<<<blocks(8L * 4 * 1024 * 144), 256, 0, stream>>>(frame, frT, 8L * 4 * 1024 * 144);
  prep_context<<<blocks((long)B * C * S), 256, 0, stream>>>(ctx, xF, xT, (long)B * C * S);

  // ---- non-local layers ----
  for (int l = 0; l < 2; ++l) {
    const __bf16* wk = wKb + (long)l * Cf * C;
    const __bf16* wq = wQb + (long)l * Cf * C;
    const __bf16* wv = wVb + (long)l * Cf * C;
    const __bf16* wo = wOb + (long)l * C * Cf;
    // K/Q/V = W * x   (M=512, N=864, K=1024)
    dim3 gKQV((S + 63) / 64, Cf / 64, B);
    gemm_nt<true, false><<<gKQV, 128, 0, stream>>>(wk, 0, C, xT, (long)S * C, C,
                                                   Kf, (long)Cf * S, S, Kb + l * Cf, Cf, S, C);
    gemm_nt<true, false><<<gKQV, 128, 0, stream>>>(wq, 0, C, xT, (long)S * C, C,
                                                   Qf, (long)Cf * S, S, Qb + l * Cf, Cf, S, C);
    gemm_nt<true, false><<<gKQV, 128, 0, stream>>>(wv, 0, C, xT, (long)S * C, C,
                                                   Vf, (long)Cf * S, S, Vb + l * Cf, Cf, S, C);
    // transpose K,Q to bf16 [S][Cf]
    t_cs<<<blocks((long)B * Cf * S), 256, 0, stream>>>(Kf, KtB, Cf, S, (long)B * Cf * S);
    t_cs<<<blocks((long)B * Cf * S), 256, 0, stream>>>(Qf, QtB, Cf, S, (long)B * Cf * S);
    // KQ[s][t] = K^T Q   (M=N=864, K=512)
    dim3 gAtt((S + 63) / 64, (S + 63) / 64, B);
    gemm_nt<false, false><<<gAtt, 128, 0, stream>>>(KtB, (long)S * Cf, Cf, QtB, (long)S * Cf, Cf,
                                                    KQ, (long)S * S, S, nullptr, S, S, Cf);
    rowsum_softmax<<<B * S, 256, 0, stream>>>(KQ, RS, S, 0.03125f);  // 1/sqrt(1024)
    vn_transpose<<<blocks((long)B * Cf * S), 256, 0, stream>>>(Vf, RS, VntB, (long)B * Cf * S);
    // O = relu(Ow * Vn + Ob)   (M=1024, N=864, K=512)
    dim3 gO((S + 63) / 64, C / 64, B);
    gemm_nt<true, true><<<gO, 128, 0, stream>>>(wo, 0, Cf, VntB, (long)S * Cf, Cf,
                                                Of, (long)C * S, S, Ob + l * C, C, S, Cf);
    residual_add<<<blocks((long)B * C * S), 256, 0, stream>>>(xF, Of, (long)B * C * S);
    bn_fused<<<C, 256, 0, stream>>>(xF, xT, gamma + l * C, beta + l * C, B, C, S);
  }

  // ---- goal path: Conv3d k=3 + relu + global mean ----
  tconv_gemm<<<dim3(7, 2, 8), 128, 0, stream>>>(tpwB, xT, tpb, gconv);
  pool_goal<<<8, 128, 0, stream>>>(gconv, out);  // goal_embed -> out[0..1023]

  // ---- frame path: two ConvTranspose2d(k=2,s=2) ----
  up1_gemm<<<dim3(72, 8, 4), 128, 0, stream>>>(up1A, frT, u1b, h1T);
  up2_gemm<<<dim3(288, 2, 4), 128, 0, stream>>>(up2A, h1T, u2b, h2T);

  // ---- goal/state attention with positional encoding ----
  sg_softmax<<<32, 256, 0, stream>>>(h2T, out, attnW);
  attn_out<<<32, 128, 0, stream>>>(h2T, attnW, out);  // out[1024..5119]
}
